// KernelSelfAttention_5059471475247
// MI455X (gfx1250) — matrix operand order repaired and checked by reading, not compiled, not run on hardware
//
#include <hip/hip_runtime.h>

typedef __attribute__((ext_vector_type(16))) _Float16 v16h;
typedef __attribute__((ext_vector_type(8)))  _Float16 v8h;
typedef __attribute__((ext_vector_type(8)))  float    v8f;
typedef int i32x4 __attribute__((vector_size(16)));   // matches builtin V4i prototype

#define NPTS  4096
#define BATCH 2

__device__ __forceinline__ float fexp2(float x) { return __builtin_amdgcn_exp2f(x); }

__device__ __forceinline__ v16h make_v16(v8h lo, v8h hi) {
    v16h r;
#pragma unroll
    for (int t = 0; t < 8; ++t) { r[t] = lo[t]; r[8 + t] = hi[t]; }
    return r;
}

// ---- CDNA5 async global->LDS copies (ASYNCcnt) -----------------------------
__device__ __forceinline__ void async_ld_b128(const void* g, void* l) {
#if __has_builtin(__builtin_amdgcn_global_load_async_to_lds_b128)
    __builtin_amdgcn_global_load_async_to_lds_b128(
        (__attribute__((address_space(1))) i32x4*)g,
        (__attribute__((address_space(3))) i32x4*)l, 0, 0);
#else
    unsigned lo = (unsigned)(unsigned long long)l;   // LDS offset lives in low 32 bits
    asm volatile("global_load_async_to_lds_b128 %0, %1, off"
                 :: "v"(lo), "v"(g) : "memory");
#endif
}

__device__ __forceinline__ void async_ld_b32(const void* g, void* l) {
#if __has_builtin(__builtin_amdgcn_global_load_async_to_lds_b32)
    __builtin_amdgcn_global_load_async_to_lds_b32(
        (__attribute__((address_space(1))) int*)g,
        (__attribute__((address_space(3))) int*)l, 0, 0);
#else
    unsigned lo = (unsigned)(unsigned long long)l;
    asm volatile("global_load_async_to_lds_b32 %0, %1, off"
                 :: "v"(lo), "v"(g) : "memory");
#endif
}

__device__ __forceinline__ void wait_async0(void) {
#if __has_builtin(__builtin_amdgcn_s_wait_asynccnt)
    __builtin_amdgcn_s_wait_asynccnt(0);
#else
    asm volatile("s_wait_asynccnt 0x0" ::: "memory");
#endif
}

// ---------------------------------------------------------------------------
// Kernel 1: weight prep.
//   Wallt[col][d] = Wv[col>>5][d][col&31]  (f16, row per output col, contig K)
//   Wot[o][k]     = Wo[k][o]               (f16)
//   bvall[col]    = bv[col]
// ---------------------------------------------------------------------------
__global__ void prep_weights(const float* __restrict__ Wv, const float* __restrict__ bv,
                             const float* __restrict__ Wo,
                             _Float16* __restrict__ Wallt, _Float16* __restrict__ Wot,
                             float* __restrict__ bvall) {
    int idx = blockIdx.x * blockDim.x + threadIdx.x;   // 0 .. 16383
    int col = idx >> 7, d = idx & 127;
    Wallt[col * 128 + d] = (_Float16)Wv[(col >> 5) * 4096 + d * 32 + (col & 31)];
    Wot[col * 128 + d]   = (_Float16)Wo[d * 128 + col];
    if (idx < 128) bvall[idx] = bv[idx];
}

// ---------------------------------------------------------------------------
// Kernel 2: V = features @ Wall + bv  via WMMA, stored transposed:
//   Vt[b][col][n]  (f16) so that attention B-tiles are contiguous in n (=K).
// ---------------------------------------------------------------------------
__global__ __launch_bounds__(128) void value_gemm(const float* __restrict__ feat,
                                                  const _Float16* __restrict__ Wallt,
                                                  const float* __restrict__ bvall,
                                                  _Float16* __restrict__ Vt) {
    __shared__ _Float16 fLDS[64 * 144];   // 64 rows, padded stride 144 halfs
    int b    = blockIdx.x / (NPTS / 64);
    int base = (blockIdx.x % (NPTS / 64)) * 64;
    int tid = threadIdx.x, lane = tid & 31, wave = tid >> 5;

    for (int r = 0; r < 64; ++r)
        fLDS[r * 144 + tid] = (_Float16)feat[((size_t)(b * NPTS + base + r)) * 128 + tid];
    __syncthreads();

    int m     = lane & 15;
    int klo   = (lane < 16) ? 0 : 8;    // A-tile per-lane K base (ISA 16-bit A layout)
    int kboff = (lane < 16) ? 0 : 16;   // unused; B tile also uses klo
    int hi8   = (lane & 16) ? 8 : 0;
    int mrow  = wave * 16 + m;

    v16h at[4];
#pragma unroll
    for (int q = 0; q < 4; ++q) {
        const v8h* p = (const v8h*)&fLDS[mrow * 144 + q * 32 + klo];
        at[q] = make_v16(p[0], p[2]);   // K {klo..klo+7} and {klo+16..klo+23}
    }

#pragma unroll
    for (int nh = 0; nh < 8; ++nh) {
        int col = nh * 16 + m;
        v8f acc = {};
#pragma unroll
        for (int q = 0; q < 4; ++q) {
            const v8h* wp = (const v8h*)&Wallt[col * 128 + q * 32 + klo];
            v16h bt = make_v16(wp[0], wp[2]);
            acc = __builtin_amdgcn_wmma_f32_16x16x32_f16(false, at[q], false, bt,
                                                         (short)0, acc, false, false);
        }
        float bvv = bvall[col];
        v8h outv;
#pragma unroll
        for (int r = 0; r < 8; ++r) outv[r] = (_Float16)(acc[r] + bvv);
        *(v8h*)&Vt[((size_t)(b * 128 + col)) * NPTS + base + wave * 16 + hi8] = outv;
    }
}

// ---------------------------------------------------------------------------
// Kernel 3: fused Gaussian attention + normalization + output projection,
// with double-buffered async global->LDS pipeline (ASYNCcnt).
// ---------------------------------------------------------------------------
__global__ __launch_bounds__(128) void attn_main(const float* __restrict__ coords,
                                                 const _Float16* __restrict__ Vt,
                                                 const _Float16* __restrict__ Wot,
                                                 const float* __restrict__ bo,
                                                 float* __restrict__ out) {
    __shared__ float4   cLDS[2][64];         // j-chunk coords (padded to float4)
    __shared__ _Float16 VLDS[2][128 * 72];   // V chunk: 128 cols x 64 j, stride 72
    __shared__ _Float16 mhLDS[64 * 144];     // normalized multi-head, stride 144

    int b    = blockIdx.x / (NPTS / 64);
    int base = (blockIdx.x % (NPTS / 64)) * 64;
    int tid = threadIdx.x, lane = tid & 31, wave = tid >> 5;
    int m     = lane & 15;
    int klo   = (lane < 16) ? 0 : 8;
    int kboff = (lane < 16) ? 0 : 16;
    int hi8   = (lane & 16) ? 8 : 0;

    int i_row = base + wave * 16 + m;
    const float* cp = coords + ((size_t)(b * NPTS + i_row)) * 3;
    float cix = cp[0], ciy = cp[1], ciz = cp[2];

    // -log2(e) / ls^2 for ls = 0.5, 1, 2, 4
    const float CH[4] = {-5.771780163555854f, -1.4426950408889634f,
                         -0.36067376022224085f, -0.09016844005556021f};

    v8f zero = {};
    v8f acc[4][2];
#pragma unroll
    for (int h = 0; h < 4; ++h) { acc[h][0] = zero; acc[h][1] = zero; }
    float rsum[4] = {0.f, 0.f, 0.f, 0.f};

    const _Float16* Vtb = Vt + (size_t)b * 128 * NPTS;

    auto issue_chunk = [&](int jb, int buf) {
        if (tid < 64) {
            const float* q = coords + ((size_t)(b * NPTS + jb + tid)) * 3;
            float* dst = (float*)&cLDS[buf][tid];
            async_ld_b32(q + 0, dst + 0);
            async_ld_b32(q + 1, dst + 1);
            async_ld_b32(q + 2, dst + 2);
        }
        const _Float16* src = Vtb + (size_t)tid * NPTS + jb;
        _Float16* dst = &VLDS[buf][tid * 72];
#pragma unroll
        for (int q8 = 0; q8 < 8; ++q8)
            async_ld_b128(src + q8 * 8, dst + q8 * 8);
    };

    issue_chunk(0, 0);                        // prologue

    for (int c = 0; c < NPTS / 64; ++c) {
        int cur = c & 1;
        wait_async0();                        // this wave's loads for buf[cur] done
        __syncthreads();                      // everyone's loads visible; prev compute done
        if ((c + 1) * 64 < NPTS)
            issue_chunk((c + 1) * 64, cur ^ 1);   // overlap next loads with compute

#pragma unroll
        for (int ks = 0; ks < 2; ++ks) {
            int koff = ks * 32;
            float d2v[16];
#pragma unroll
            for (int s = 0; s < 16; ++s) {
                int K = koff + klo + (s < 8 ? s : s + 8);
                float4 cj = cLDS[cur][K];
                float dx = cix - cj.x, dy = ciy - cj.y, dz = ciz - cj.z;
                d2v[s] = dx * dx + dy * dy + dz * dz;
            }
#pragma unroll
            for (int h = 0; h < 4; ++h) {
                v16h a; float lsm = 0.f;
#pragma unroll
                for (int s = 0; s < 16; ++s) {
                    float w = fexp2(d2v[s] * CH[h]);
                    lsm += w;
                    a[s] = (_Float16)w;
                }
                rsum[h] += lsm;
#pragma unroll
                for (int nh = 0; nh < 2; ++nh) {
                    const v8h* vp =
                        (const v8h*)&VLDS[cur][(h * 32 + nh * 16 + m) * 72 + koff + klo];
                    v16h bt = make_v16(vp[0], vp[2]);
                    acc[h][nh] = __builtin_amdgcn_wmma_f32_16x16x32_f16(
                        false, a, false, bt, (short)0, acc[h][nh], false, false);
                }
            }
        }
    }
    __syncthreads();

    // row sums: lane L and L+16 own disjoint K -> xor-16 reduce, then broadcast per row
#pragma unroll
    for (int h = 0; h < 4; ++h) {
        float tot = rsum[h] + __shfl_xor(rsum[h], 16);
#pragma unroll
        for (int r = 0; r < 8; ++r) {
            float den = __shfl(tot, r + hi8);        // lane r+hi8 holds row r+hi8 total
            float inv = 1.0f / (den + 1e-8f);
#pragma unroll
            for (int nh = 0; nh < 2; ++nh)
                mhLDS[(wave * 16 + r + hi8) * 144 + h * 32 + nh * 16 + m] =
                    (_Float16)(acc[h][nh][r] * inv);
        }
    }
    __syncthreads();

    // fused output projection: out = mh @ Wo + bo
    v16h at[4];
    int mrow = wave * 16 + m;
#pragma unroll
    for (int q = 0; q < 4; ++q) {
        const v8h* p = (const v8h*)&mhLDS[mrow * 144 + q * 32 + klo];
        at[q] = make_v16(p[0], p[2]);
    }
#pragma unroll
    for (int nh = 0; nh < 8; ++nh) {
        int ocol = nh * 16 + m;
        v8f oacc = {};
#pragma unroll
        for (int q = 0; q < 4; ++q) {
            const v8h* wp = (const v8h*)&Wot[ocol * 128 + q * 32 + klo];
            v16h bt = make_v16(wp[0], wp[2]);
            oacc = __builtin_amdgcn_wmma_f32_16x16x32_f16(false, at[q], false, bt,
                                                          (short)0, oacc, false, false);
        }
        float bov = bo[ocol];
#pragma unroll
        for (int r = 0; r < 8; ++r)
            out[((size_t)(b * NPTS + base + wave * 16 + r + hi8)) * 128 + ocol] =
                oacc[r] + bov;
    }
}

// ---------------------------------------------------------------------------
extern "C" void kernel_launch(void* const* d_in, const int* in_sizes, int n_in,
                              void* d_out, int out_size, void* d_ws, size_t ws_size,
                              hipStream_t stream) {
    (void)in_sizes; (void)n_in; (void)out_size; (void)ws_size;
    const float* features = (const float*)d_in[0];
    const float* coords   = (const float*)d_in[1];
    const float* Wv       = (const float*)d_in[2];
    const float* bv       = (const float*)d_in[3];
    const float* Wo       = (const float*)d_in[4];
    const float* bo       = (const float*)d_in[5];
    float* out = (float*)d_out;

    char* ws = (char*)d_ws;
    const size_t VT_BYTES = (size_t)BATCH * 128 * NPTS * sizeof(_Float16); // 2 MiB
    _Float16* Vt    = (_Float16*)ws;
    _Float16* Wallt = (_Float16*)(ws + VT_BYTES);
    _Float16* Wot   = (_Float16*)(ws + VT_BYTES + 128 * 128 * sizeof(_Float16));
    float*    bvall = (float*)   (ws + VT_BYTES + 2 * 128 * 128 * sizeof(_Float16));

    prep_weights<<<64, 256, 0, stream>>>(Wv, bv, Wo, Wallt, Wot, bvall);
    value_gemm<<<BATCH * (NPTS / 64), 128, 0, stream>>>(features, Wallt, bvall, Vt);
    attn_main<<<BATCH * (NPTS / 64), 128, 0, stream>>>(coords, Vt, Wot, bo, out);
}
